// ValueHead_77644418777370
// MI455X (gfx1250) — matrix operand order repaired and checked by reading, not compiled, not run on hardware
//
#include <hip/hip_runtime.h>

// ---------------------------------------------------------------------------
// CDNA5 (gfx1250) fused ValueHead: 4-group tiny attention + MLP, all GEMMs on
// v_wmma_f32_16x16x32_bf16 (wave32). One wave handles 16 batch rows end-to-end.
// Input staging uses the gfx1250 async global->LDS copy path (ASYNCcnt).
// B fragments are reused across the 3 attention words to cut L2 traffic 3x.
// ---------------------------------------------------------------------------

typedef __attribute__((ext_vector_type(16))) __bf16 v16bf;
typedef __attribute__((ext_vector_type(8)))  float  v8f;

#define WMMA_BF16(A, B, C) \
  __builtin_amdgcn_wmma_f32_16x16x32_bf16(false, (A), false, (B), (short)0, (C), false, false)

constexpr int FEAT = 384;   // sum(GROUP_DIMS)
constexpr int NW   = 3;     // words
constexpr int TB   = 16;    // batch rows per wave (WMMA M)

// Generic pointers to __shared__ carry the LDS byte offset in their low 32 bits
// (ISA 10.2: LDS aperture maps ADDR[31:0] -> LDS offset).
__device__ __forceinline__ unsigned lds_off(const void* p) {
  return (unsigned)(unsigned long long)p;
}

// Async 16B/lane copy: LDS[vdst] = MEM[vaddr]; tracked by ASYNCcnt.
__device__ __forceinline__ void async_copy_b128(unsigned lds_byte, const void* gptr) {
  asm volatile("global_load_async_to_lds_b128 %0, %1, off"
               :: "v"(lds_byte), "v"((unsigned long long)gptr)
               : "memory");
}
__device__ __forceinline__ void wait_async0() {
  asm volatile("s_wait_asynccnt 0" ::: "memory");
}

__device__ __forceinline__ unsigned pack2bf(float x, float y) {
  union { __bf16 h[2]; unsigned u; } t;
  t.h[0] = (__bf16)x; t.h[1] = (__bf16)y;
  return t.u;   // v_cvt_pk_bf16_f32
}

// ---- A fragment (16x32 bf16, doc 7.12.2): lane = {hi=lane>>4, lo=lane&15},
// M = lo; dword j holds K pair kbase + (j>>2)*16 + hi*8 + (j&3)*2.
__device__ __forceinline__ v16bf load_a_frag(const __bf16* row, int kbase, int hi) {
  union { v16bf v; unsigned u[8]; } a;
#pragma unroll
  for (int j = 0; j < 8; ++j) {
    int k = kbase + ((j >> 2) << 4) + (hi << 3) + ((j & 3) << 1);
    a.u[j] = *reinterpret_cast<const unsigned*>(row + k);
  }
  return a.v;
}
// f32-source variant (row of f32 in LDS, convert while building):
__device__ __forceinline__ v16bf load_a_frag_f32(const float* row, int kbase, int hi) {
  union { v16bf v; unsigned u[8]; } a;
#pragma unroll
  for (int j = 0; j < 8; ++j) {
    int k = kbase + ((j >> 2) << 4) + (hi << 3) + ((j & 3) << 1);
    float2 p = *reinterpret_cast<const float2*>(row + k);
    a.u[j] = pack2bf(p.x, p.y);
  }
  return a.v;
}

// ---- B fragment (32x16 bf16): packed tiles, 512 bf16 per tile, lane-major ----
__device__ __forceinline__ v16bf load_b_frag(const __bf16* base, int tile, int lane) {
  return *reinterpret_cast<const v16bf*>(base + tile * 512 + lane * 16);
}

// ---------------------------------------------------------------------------
// Kernel 0: repack a [K x N] row-major fp32 weight into bf16 B-fragment tiles.
// ---------------------------------------------------------------------------
__global__ void pack_b_kernel(const float* __restrict__ Wm, int K, int N,
                              __bf16* __restrict__ out) {
  int tid = blockIdx.x * blockDim.x + threadIdx.x;
  int total = (K >> 5) * (N >> 4) * 512;
  if (tid >= total) return;
  int e    = tid & 15;
  int lane = (tid >> 4) & 31;
  int tile = tid >> 9;
  int ntn  = N >> 4;
  int nt = tile % ntn, kt = tile / ntn;
  int k = (kt << 5) + ((e >> 3) << 4) + ((lane >> 4) << 3) + (e & 7);
  int n = (nt << 4) + (lane & 15);
  out[tid] = (__bf16)Wm[k * N + n];
}

// ---------------------------------------------------------------------------
// Fused main kernel: blockDim = 32 (one wave), block b handles rows [16b,16b+16)
// ---------------------------------------------------------------------------
__global__ __launch_bounds__(32)
void valuehead_fused(const float* __restrict__ words,
                     const float* __restrict__ bq0, const float* __restrict__ bo0,
                     const float* __restrict__ bq1, const float* __restrict__ bo1,
                     const float* __restrict__ bq2, const float* __restrict__ bo2,
                     const float* __restrict__ bq3, const float* __restrict__ bo3,
                     const float* __restrict__ bi,  const float* __restrict__ wout,
                     const float* __restrict__ bout,
                     const __bf16* __restrict__ wpk,
                     float* __restrict__ out) {
  // region A: X (f32, async-staged, 24KB) -> QKV (bf16, 36.8KB) -> inner (f32 16KB)
  __shared__ __align__(16) char   smemA[NW * TB * 384 * 2];   // 36864 B
  __shared__ __align__(16) __bf16 smemAV[NW * TB * 128];      // 12288 B
  __shared__ __align__(8)  __bf16 smemStage[NW * TB * 64];    //  6144 B
  __shared__ float smemAttn[TB * 9];                          //   576 B

  const int lane = (int)threadIdx.x;
  const int lo = lane & 15, hi = lane >> 4;
  const int b0 = (int)blockIdx.x * TB;

  constexpr int GDv[4]   = {128, 64, 128, 64};
  constexpr int GOFFv[4] = {0, 128, 192, 320};
  constexpr int WSQ[4]   = {0, 49152, 61440, 110592};          // packed wqkv_g
  constexpr int WSO[4]   = {122880, 131072, 135168, 143360};   // packed wo_g
  constexpr int WSI      = 147456;                             // packed wi
  const float* bqp[4] = {bq0, bq1, bq2, bq3};
  const float* bop[4] = {bo0, bo1, bo2, bo3};

  v16bf actf[24];   // register-resident MLP A fragments (768 = 24 K-steps)

#pragma unroll
  for (int g = 0; g < 4; ++g) {
    const int d   = GDv[g];
    const int off = GOFFv[g];
    const int KT  = d >> 5;          // K tiles for QKV / proj (4 or 2)
    const int NT3 = (3 * d) >> 4;    // N tiles of QKV output (24 or 12)
    const int S3  = 3 * d;
    const int d4  = d >> 2;          // 16B chunks per row

    // ---- async-stage X[g] (f32) into LDS, layout [w][m][c], 16B per lane ----
    float* Xf = reinterpret_cast<float*>(smemA);
    {
      const unsigned xbase = lds_off(Xf);
      const int tot = TB * NW * d4;  // 16B chunks, divisible by 32
      for (int idx = lane; idx < tot; idx += 32) {
        int row = idx / d4;          // = w*TB + m
        int c4  = idx - row * d4;
        int w   = row >> 4;
        int m   = row & 15;
        const float* src = words + (b0 + m) * (NW * FEAT) + w * FEAT + off + c4 * 4;
        async_copy_b128(xbase + (unsigned)((row * d + c4 * 4) * 4), src);
      }
    }
    wait_async0();
    __syncthreads();

    // ---- build A fragments for X (kept in VGPRs across all N tiles) ----
    v16bf aq[3][4];
#pragma unroll
    for (int w = 0; w < NW; ++w)
#pragma unroll
      for (int kt = 0; kt < 4; ++kt)
        if (kt < KT) aq[w][kt] = load_a_frag_f32(Xf + (w * TB + lo) * d, kt * 32, hi);
    __syncthreads();

    // ---- QKV GEMM: B tiles loaded once per nt, reused by all 3 words ----
    __bf16* QKV = reinterpret_cast<__bf16*>(smemA);   // [w][m][3d] (X dead)
    const __bf16* wq = wpk + WSQ[g];
    const float*  bq = bqp[g];
    for (int nt = 0; nt < NT3; ++nt) {
      v16bf bf[4];
#pragma unroll
      for (int kt = 0; kt < 4; ++kt)
        if (kt < KT) bf[kt] = load_b_frag(wq, kt * NT3 + nt, lane);
      float bias = bq[nt * 16 + lo];
#pragma unroll
      for (int w = 0; w < NW; ++w) {
        v8f acc = {};
#pragma unroll
        for (int kt = 0; kt < 4; ++kt)
          if (kt < KT) acc = WMMA_BF16(aq[w][kt], bf[kt], acc);
#pragma unroll
        for (int r8 = 0; r8 < 8; ++r8) {               // D: m = r8 + 8*hi, n = lo
          int m = r8 + 8 * hi;
          QKV[(w * TB + m) * S3 + nt * 16 + lo] = (__bf16)(acc[r8] + bias);
        }
      }
    }
    __syncthreads();

    // ---- 3x3 scores: all 32 lanes (row lo, K-half hi), pairwise bf16 loads ----
    {
      const int m = lo;
      const int cbase = hi * (d >> 1);
      float s00 = 0, s01 = 0, s02 = 0, s10 = 0, s11 = 0, s12 = 0, s20 = 0, s21 = 0, s22 = 0;
      for (int c = 0; c < (d >> 1); c += 2) {
        int cc = cbase + c;
        union { unsigned u; __bf16 h[2]; } q0, q1, q2, k0, k1, k2;
        q0.u = *reinterpret_cast<const unsigned*>(QKV + (0 * TB + m) * S3 + cc);
        q1.u = *reinterpret_cast<const unsigned*>(QKV + (1 * TB + m) * S3 + cc);
        q2.u = *reinterpret_cast<const unsigned*>(QKV + (2 * TB + m) * S3 + cc);
        k0.u = *reinterpret_cast<const unsigned*>(QKV + (0 * TB + m) * S3 + d + cc);
        k1.u = *reinterpret_cast<const unsigned*>(QKV + (1 * TB + m) * S3 + d + cc);
        k2.u = *reinterpret_cast<const unsigned*>(QKV + (2 * TB + m) * S3 + d + cc);
        float qa0 = (float)q0.h[0], qb0 = (float)q0.h[1];
        float qa1 = (float)q1.h[0], qb1 = (float)q1.h[1];
        float qa2 = (float)q2.h[0], qb2 = (float)q2.h[1];
        float ka0 = (float)k0.h[0], kb0 = (float)k0.h[1];
        float ka1 = (float)k1.h[0], kb1 = (float)k1.h[1];
        float ka2 = (float)k2.h[0], kb2 = (float)k2.h[1];
        s00 += qa0 * ka0 + qb0 * kb0; s01 += qa0 * ka1 + qb0 * kb1; s02 += qa0 * ka2 + qb0 * kb2;
        s10 += qa1 * ka0 + qb1 * kb0; s11 += qa1 * ka1 + qb1 * kb1; s12 += qa1 * ka2 + qb1 * kb2;
        s20 += qa2 * ka0 + qb2 * kb0; s21 += qa2 * ka1 + qb2 * kb1; s22 += qa2 * ka2 + qb2 * kb2;
      }
      float s[3][3] = {{s00, s01, s02}, {s10, s11, s12}, {s20, s21, s22}};
#pragma unroll
      for (int i = 0; i < 3; ++i)
#pragma unroll
        for (int j = 0; j < 3; ++j) s[i][j] += __shfl_xor(s[i][j], 16, 32);
      if (hi == 0) {
        const float scale = (d == 128) ? 0.08838834764831845f : 0.125f;  // 1/sqrt(d)
#pragma unroll
        for (int i = 0; i < 3; ++i) {
          float a0 = s[i][0] * scale, a1 = s[i][1] * scale, a2 = s[i][2] * scale;
          float mx = fmaxf(a0, fmaxf(a1, a2));
          float e0 = __expf(a0 - mx), e1 = __expf(a1 - mx), e2 = __expf(a2 - mx);
          float inv = 1.0f / (e0 + e1 + e2);
          smemAttn[m * 9 + i * 3 + 0] = e0 * inv;
          smemAttn[m * 9 + i * 3 + 1] = e1 * inv;
          smemAttn[m * 9 + i * 3 + 2] = e2 * inv;
        }
      }
    }
    __syncthreads();

    // ---- attn @ V: all 32 lanes, lane owns (row lo, half hi of channels) ----
    {
      const int m = lo;
      const int c0 = hi * (d >> 1);
      float a[3][3];
#pragma unroll
      for (int i = 0; i < 3; ++i)
#pragma unroll
        for (int j = 0; j < 3; ++j) a[i][j] = smemAttn[m * 9 + i * 3 + j];
      for (int c = 0; c < (d >> 1); ++c) {
        int cc = c0 + c;
        float v0 = (float)QKV[(0 * TB + m) * S3 + 2 * d + cc];
        float v1 = (float)QKV[(1 * TB + m) * S3 + 2 * d + cc];
        float v2 = (float)QKV[(2 * TB + m) * S3 + 2 * d + cc];
#pragma unroll
        for (int i = 0; i < 3; ++i)
          smemAV[(i * TB + m) * d + cc] =
              (__bf16)(a[i][0] * v0 + a[i][1] * v1 + a[i][2] * v2);
      }
    }
    __syncthreads();

    // ---- projection: B tiles loaded once per nt, reused by all 3 words ----
    const __bf16* wo = wpk + WSO[g];
    const float*  bo = bop[g];
    {
      v16bf av[3][4];
#pragma unroll
      for (int w = 0; w < NW; ++w)
#pragma unroll
        for (int kt = 0; kt < 4; ++kt)
          if (kt < KT) av[w][kt] = load_a_frag(smemAV + (w * TB + lo) * d, kt * 32, hi);
#pragma unroll
      for (int nt = 0; nt < 4; ++nt) {
        v16bf bf[4];
#pragma unroll
        for (int kt = 0; kt < 4; ++kt)
          if (kt < KT) bf[kt] = load_b_frag(wo, kt * 4 + nt, lane);
        float bias = bo[nt * 16 + lo];
#pragma unroll
        for (int w = 0; w < NW; ++w) {
          v8f acc = {};
#pragma unroll
          for (int kt = 0; kt < 4; ++kt)
            if (kt < KT) acc = WMMA_BF16(av[w][kt], bf[kt], acc);
#pragma unroll
          for (int r8 = 0; r8 < 8; ++r8) {
            int m = r8 + 8 * hi;
            float v = acc[r8] + bias;
            smemStage[(w * TB + m) * 64 + nt * 16 + lo] = (__bf16)(v > 0.f ? v : 0.f);
          }
        }
      }
    }
    __syncthreads();
    // flat k range for (g,w) is [(3g+w)*64, +64) -> MLP K tiles (3g+w)*2 + t
#pragma unroll
    for (int w = 0; w < NW; ++w)
#pragma unroll
      for (int t = 0; t < 2; ++t)
        actf[(3 * g + w) * 2 + t] =
            load_a_frag(smemStage + (w * TB + lo) * 64, t * 32, hi);
    __syncthreads();
  } // groups

  // ---- MLP: [16 x 768] @ wi[768 x 256], relu -> inner (f32 in LDS) ----
  float* inner = reinterpret_cast<float*>(smemA);   // [16][256]
  const __bf16* wip = wpk + WSI;
  for (int nt = 0; nt < 16; ++nt) {
    v8f acc = {};
#pragma unroll
    for (int kt = 0; kt < 24; ++kt)
      acc = WMMA_BF16(actf[kt], load_b_frag(wip, kt * 16 + nt, lane), acc);
    float bias = bi[nt * 16 + lo];
#pragma unroll
    for (int r8 = 0; r8 < 8; ++r8) {
      int m = r8 + 8 * hi;
      float v = acc[r8] + bias;
      inner[m * 256 + nt * 16 + lo] = v > 0.f ? v : 0.f;
    }
  }
  __syncthreads();

  // ---- final head: inner[16 x 256] @ wout[256] + bout, split across halves ----
  {
    const int m = lo;
    float sum = 0.f;
    for (int c = hi * 128; c < hi * 128 + 128; ++c)
      sum += inner[m * 256 + c] * wout[c];
    sum += __shfl_xor(sum, 16, 32);
    if (hi == 0) out[b0 + m] = sum + bout[0];
  }
}

// ---------------------------------------------------------------------------
// Host launcher
// ---------------------------------------------------------------------------
extern "C" void kernel_launch(void* const* d_in, const int* in_sizes, int n_in,
                              void* d_out, int out_size, void* d_ws, size_t ws_size,
                              hipStream_t stream) {
  (void)n_in; (void)out_size; (void)ws_size;
  __bf16* wpk = (__bf16*)d_ws;   // ~688 KB of packed bf16 weights

  struct Job { int idx, K, N, off; };
  const Job jobs[9] = {
      {1, 128, 384, 0},      {5, 64, 192, 49152},
      {9, 128, 384, 61440},  {13, 64, 192, 110592},
      {3, 128, 64, 122880},  {7, 64, 64, 131072},
      {11, 128, 64, 135168}, {15, 64, 64, 143360},
      {17, 768, 256, 147456}};
  for (int j = 0; j < 9; ++j) {
    int total = (jobs[j].K / 32) * (jobs[j].N / 16) * 512;
    pack_b_kernel<<<(total + 255) / 256, 256, 0, stream>>>(
        (const float*)d_in[jobs[j].idx], jobs[j].K, jobs[j].N, wpk + jobs[j].off);
  }

  const int B = in_sizes[0] / (NW * FEAT);   // 65536
  valuehead_fused<<<B / TB, 32, 0, stream>>>(
      (const float*)d_in[0],
      (const float*)d_in[2],  (const float*)d_in[4],
      (const float*)d_in[6],  (const float*)d_in[8],
      (const float*)d_in[10], (const float*)d_in[12],
      (const float*)d_in[14], (const float*)d_in[16],
      (const float*)d_in[18], (const float*)d_in[19], (const float*)d_in[20],
      wpk, (float*)d_out);
}
